// AetherSparcNet_21792664060793
// MI455X (gfx1250) — matrix operand order repaired and checked by reading, not compiled, not run on hardware
//
#include <hip/hip_runtime.h>
#include <hip/hip_bf16.h>
#include <math.h>

typedef __attribute__((ext_vector_type(16))) _Float16 v16h;
typedef __attribute__((ext_vector_type(2)))  _Float16 v2h;
typedef __attribute__((ext_vector_type(8)))  float    v8f;

#define NROWS    1048576
#define HDIM     128
#define THRESH   0.045f
#define TAUC     20.0f
#define CHUNK    4096              // rows per block in scan kernels
#define NBLK     (NROWS / CHUNK)   // 256
#define ROWS_PER_BLOCK 256         // 8 waves x 32 rows

// ---------------------------------------------------------------------------
// Kernel 1: fused MLP.  256 threads = 8 waves; each wave owns TWO 16-row A
// tiles (32 rows) so every B-fragment LDS load feeds two v_wmma ops
// (1 ds_load_b128 per wmma).  W2^T is staged in LDS pre-packed into f16 WMMA
// B-fragment order.  h1 = relu(x*W1+b1) is built directly in the A-fragment
// register layout; layer 3 (dot with W3) folds into the epilogue via shfl.
// ---------------------------------------------------------------------------
__global__ __launch_bounds__(256) void fused_mlp_kernel(
    const float* __restrict__ x,  const float* __restrict__ W1,
    const float* __restrict__ b1, const float* __restrict__ W2,
    const float* __restrict__ b2, const float* __restrict__ W3,
    const float* __restrict__ b3, float* __restrict__ y)
{
    // packed B fragments: [nt(8)][kc(4)][lane(32)][t(16)] halves = 16384 = 32KB
    __shared__ _Float16 sBp[8 * 4 * 32 * 16];
    __shared__ float sW1[HDIM], sb1[HDIM], sb2[HDIM], sW3[HDIM];

    const int tid = threadIdx.x;
    if (tid < HDIM) {
        sW1[tid] = W1[tid];
        sb1[tid] = b1[tid];
        sb2[tid] = b2[tid];
        sW3[tid] = W3[tid];
    }
    // Pack W2^T (B[k][j] = W2[j*128+k]) into WMMA B-fragment order, two
    // halves at a time (t=2v,2v+1 -> contiguous k in a W2 row).
    // B 32x16, contiguous-k order used here (not the ISA table order): lanes 0-15 hold K=0..15 (VGPR v -> K=2v,2v+1),
    // lanes 16-31 hold K=16..31.
    for (int idx2 = tid; idx2 < 8 * 4 * 32 * 8; idx2 += 256) {
        const int v  = idx2 & 7;
        const int L  = (idx2 >> 3) & 31;
        const int kc = (idx2 >> 8) & 3;
        const int nt = idx2 >> 10;
        const int k0 = kc * 32 + ((L < 16) ? (2 * v) : (16 + 2 * v));
        const int j  = nt * 16 + (L & 15);
        const float w0 = W2[j * HDIM + k0];
        const float w1 = W2[j * HDIM + k0 + 1];
        v2h pk;
        pk[0] = (_Float16)w0;
        pk[1] = (_Float16)w1;
        ((v2h*)sBp)[idx2] = pk;
    }
    __syncthreads();

    const int wave = tid >> 5;
    const int lane = tid & 31;
    const int m    = lane & 15;            // row within a 16-row tile
    const int r0   = blockIdx.x * ROWS_PER_BLOCK + wave * 32;

    const float xv0 = x[r0 + m];           // tile 0 row input
    const float xv1 = x[r0 + 16 + m];      // tile 1 row input

    v8f acc0[8], acc1[8];
    #pragma unroll
    for (int nt = 0; nt < 8; ++nt) { acc0[nt] = (v8f){}; acc1[nt] = (v8f){}; }

    // K loop: 128 = 4 chunks of 32
    #pragma unroll
    for (int kc = 0; kc < 4; ++kc) {
        // A fragments (16x32 f16): lane<16 holds K {0..15},
        // lane>=16 holds K {16..31} of its row.
        v16h a0, a1;
        const int kb = kc * 32 + ((lane >= 16) ? 16 : 0);
        #pragma unroll
        for (int t = 0; t < 16; ++t) {
            const int k = kb + t;
            const float w1k = sW1[k];
            const float b1k = sb1[k];
            float h0 = fmaf(xv0, w1k, b1k);
            float h1 = fmaf(xv1, w1k, b1k);
            h0 = h0 > 0.0f ? h0 : 0.0f;    // relu (layer 1)
            h1 = h1 > 0.0f ? h1 : 0.0f;
            a0[t] = (_Float16)h0;
            a1[t] = (_Float16)h1;
        }
        #pragma unroll
        for (int nt = 0; nt < 8; ++nt) {
            const v16h bfrag =
                *(const v16h*)&sBp[(((nt * 4 + kc) * 32) + lane) * 16];
            acc0[nt] = __builtin_amdgcn_wmma_f32_16x16x32_f16(
                false, a0, false, bfrag, (short)0, acc0[nt], false, false);
            acc1[nt] = __builtin_amdgcn_wmma_f32_16x16x32_f16(
                false, a1, false, bfrag, (short)0, acc1[nt], false, false);
        }
    }

    // Epilogue: h2 = relu(acc + b2); y_row = h2 . W3 + b3.
    // C/D layout: vgpr r holds row (r + 8*(lane>=16)), column nt*16 + (lane&15).
    float p0[8], p1[8];
    #pragma unroll
    for (int r = 0; r < 8; ++r) { p0[r] = 0.0f; p1[r] = 0.0f; }

    #pragma unroll
    for (int nt = 0; nt < 8; ++nt) {
        const float w3 = sW3[nt * 16 + m];
        const float bb = sb2[nt * 16 + m];
        #pragma unroll
        for (int r = 0; r < 8; ++r) {
            float h2a = acc0[nt][r] + bb;
            float h2b = acc1[nt][r] + bb;
            h2a = h2a > 0.0f ? h2a : 0.0f; // relu (layer 2)
            h2b = h2b > 0.0f ? h2b : 0.0f;
            p0[r] = fmaf(h2a, w3, p0[r]);
            p1[r] = fmaf(h2b, w3, p1[r]);
        }
    }
    // Reduce the 128-wide dot across the 16 columns held by each half-wave.
    // xor masks 1,2,4,8 never cross the 16-lane boundary on wave32.
    #pragma unroll
    for (int off = 1; off < 16; off <<= 1) {
        #pragma unroll
        for (int r = 0; r < 8; ++r) {
            p0[r] += __shfl_xor(p0[r], off, 32);
            p1[r] += __shfl_xor(p1[r], off, 32);
        }
    }
    if (m == 0) {
        const float bb3 = b3[0];
        const int base = r0 + ((lane >= 16) ? 8 : 0);
        #pragma unroll
        for (int r = 0; r < 8; ++r) {
            y[base + r]      = p0[r] + bb3;
            y[base + 16 + r] = p1[r] + bb3;
        }
    }
}

// ---------------------------------------------------------------------------
// Kernel 2: per-block max of masked time index + active count.
// ---------------------------------------------------------------------------
__global__ __launch_bounds__(256) void scan_pass1_kernel(
    const float* __restrict__ x, float* __restrict__ blockMax,
    int* __restrict__ blockCnt)
{
    __shared__ float sMax[256];
    __shared__ int   sCnt[256];
    const int tid  = threadIdx.x;
    const int base = blockIdx.x * CHUNK + tid * 16;

    float mx  = 0.0f;            // neutral matches where(mask, t, 0.0)
    int   cnt = 0;
    float prev = (base > 0) ? x[base - 1] : 0.0f;
    #pragma unroll
    for (int k = 0; k < 16; ++k) {
        const int i = base + k;
        const float cur = x[i];
        const bool mask = (i == 0) || (fabsf(cur - prev) > THRESH);
        if (mask) { mx = (float)i; ++cnt; }
        prev = cur;
    }
    sMax[tid] = mx;
    sCnt[tid] = cnt;
    __syncthreads();
    for (int s = 128; s > 0; s >>= 1) {
        if (tid < s) {
            sMax[tid] = fmaxf(sMax[tid], sMax[tid + s]);
            sCnt[tid] += sCnt[tid + s];
        }
        __syncthreads();
    }
    if (tid == 0) {
        blockMax[blockIdx.x] = sMax[0];
        blockCnt[blockIdx.x] = sCnt[0];
    }
}

// ---------------------------------------------------------------------------
// Kernel 3 (single block): exclusive prefix-max over block maxima, total count.
// ---------------------------------------------------------------------------
__global__ __launch_bounds__(256) void scan_pass2_kernel(
    const float* __restrict__ blockMax, const int* __restrict__ blockCnt,
    float* __restrict__ prefixMax, float* __restrict__ out_nactive)
{
    __shared__ float s[256];
    __shared__ int   c[256];
    const int tid = threadIdx.x;
    s[tid] = (tid < NBLK) ? blockMax[tid] : 0.0f;
    c[tid] = (tid < NBLK) ? blockCnt[tid] : 0;
    __syncthreads();
    for (int off = 1; off < 256; off <<= 1) {
        const float v  = (tid >= off) ? s[tid - off] : 0.0f;
        const int   cv = (tid >= off) ? c[tid - off] : 0;
        __syncthreads();
        s[tid] = fmaxf(s[tid], v);
        c[tid] += cv;
        __syncthreads();
    }
    if (tid < NBLK)
        prefixMax[tid] = (tid == 0) ? 0.0f : s[tid - 1];
    if (tid == 255 && out_nactive != nullptr)
        out_nactive[0] = (float)c[255];
}

// ---------------------------------------------------------------------------
// Kernel 4: finish the cummax inside each block, gather y + decay.
// ---------------------------------------------------------------------------
__global__ __launch_bounds__(256) void finalize_kernel(
    const float* __restrict__ x, const float* __restrict__ y,
    const float* __restrict__ prefixMax, float* __restrict__ out)
{
    __shared__ float sMax[256];
    const int tid  = threadIdx.x;
    const int b    = blockIdx.x;
    const int base = b * CHUNK + tid * 16;

    // pass A: thread-local max of masked t
    float mx = 0.0f;
    float prev = (base > 0) ? x[base - 1] : 0.0f;
    #pragma unroll
    for (int k = 0; k < 16; ++k) {
        const int i = base + k;
        const float cur = x[i];
        if ((i == 0) || (fabsf(cur - prev) > THRESH)) mx = (float)i;
        prev = cur;
    }
    sMax[tid] = mx;
    __syncthreads();
    // inclusive Hillis-Steele max-scan over threads
    for (int off = 1; off < 256; off <<= 1) {
        const float v = (tid >= off) ? sMax[tid - off] : 0.0f;
        __syncthreads();
        sMax[tid] = fmaxf(sMax[tid], v);
        __syncthreads();
    }
    float running = prefixMax[b];
    if (tid > 0) running = fmaxf(running, sMax[tid - 1]);

    // pass B: serial cummax over this thread's 16 rows + gather/decay
    prev = (base > 0) ? x[base - 1] : 0.0f;
    #pragma unroll
    for (int k = 0; k < 16; ++k) {
        const int i = base + k;
        const float cur = x[i];
        if ((i == 0) || (fabsf(cur - prev) > THRESH)) running = (float)i;
        const int li = (int)running;
        out[i] = y[li] * expf(-((float)i - running) / TAUC);
        prev = cur;
    }
}

// ---------------------------------------------------------------------------
extern "C" void kernel_launch(void* const* d_in, const int* in_sizes, int n_in,
                              void* d_out, int out_size, void* d_ws, size_t ws_size,
                              hipStream_t stream) {
    const float* x  = (const float*)d_in[0];
    const float* W1 = (const float*)d_in[1];
    const float* b1 = (const float*)d_in[2];
    const float* W2 = (const float*)d_in[3];
    const float* b2 = (const float*)d_in[4];
    const float* W3 = (const float*)d_in[5];
    const float* b3 = (const float*)d_in[6];
    float* outf = (float*)d_out;

    // workspace layout
    float* yBuf      = (float*)d_ws;            // NROWS floats
    float* blockMax  = yBuf + NROWS;            // NBLK floats
    int*   blockCnt  = (int*)(blockMax + NBLK); // NBLK ints
    float* prefixMax = (float*)(blockCnt + NBLK);

    float* out_nactive = (out_size > NROWS) ? (outf + NROWS) : nullptr;

    fused_mlp_kernel<<<NROWS / ROWS_PER_BLOCK, 256, 0, stream>>>(
        x, W1, b1, W2, b2, W3, b3, yBuf);
    scan_pass1_kernel<<<NBLK, 256, 0, stream>>>(x, blockMax, blockCnt);
    scan_pass2_kernel<<<1, 256, 0, stream>>>(blockMax, blockCnt, prefixMax, out_nactive);
    finalize_kernel<<<NBLK, 256, 0, stream>>>(x, yBuf, prefixMax, outf);
}
